// NNConvWrapper_90417651516149
// MI455X (gfx1250) — matrix operand order repaired and checked by reading, not compiled, not run on hardware
//
#include <hip/hip_runtime.h>
#include <math.h>

// ---------------------------------------------------------------------------
// NNConv + GAT-style attention, fused for MI455X (gfx1250, wave32, WMMA).
//
// Dominant cost: h[E,128] @ W2[128,2048]  (52.4 GFLOP fp32-equivalent).
// Strategy: per-16-edge tile, run the GEMM with v_wmma_f32_16x16x32_bf16
// using a 3-term bf16 split (hi*hi + hi*lo + lo*hi ~ fp32 accuracy) and
// consume each 16x16 output tile immediately (never materialize w to HBM).
// ---------------------------------------------------------------------------

#define N_NODES 25000
#define N_EDGES 100000
#define V_IN    32
#define V_OUT   64
#define E_INF   16
#define HID     128
#define NHEAD   8
#define CH      8
#define W2COLS  2048   // V_IN*V_OUT

typedef __bf16 bf16_t;
typedef bf16_t v16bf __attribute__((ext_vector_type(16)));
typedef float  v8f   __attribute__((ext_vector_type(8)));

// workspace layout (all offsets 256B aligned); total ~37.9 MB
#define OFF_W2HI  ((size_t)0)          // 262144 bf16 = 512 KB (swizzled B frags)
#define OFF_W2LO  ((size_t)524288)     // 512 KB
#define OFF_XJ    ((size_t)1048576)    // E*64 f32 = 25.6 MB
#define OFF_ALPHA ((size_t)26648576)   // E*8 f32  = 3.2 MB (raw alpha, then exp)
#define OFF_MKEY  ((size_t)29848576)   // N*8 u32  = 800 KB (segment max keys)
#define OFF_S     ((size_t)30648576)   // N*8 f32  = 800 KB (segment exp-sums)
#define OFF_AGG   ((size_t)31448576)   // N*64 f32 = 6.4 MB (message accumulator)

#define NEG_KEY 0x00800000u  // order-preserving encode of -FLT_MAX

__device__ __forceinline__ unsigned fkey(float f) {
    unsigned u = __float_as_uint(f);
    return (u & 0x80000000u) ? ~u : (u | 0x80000000u);
}
__device__ __forceinline__ float kinv(unsigned k) {
    unsigned u = (k & 0x80000000u) ? (k ^ 0x80000000u) : ~k;
    return __uint_as_float(u);
}

// --- Pre-swizzle W2 into WMMA B-fragment order, split into bf16 hi/lo -------
// B (32x16 bf16) fragment layout: lane l holds column N=l&15; element j of the
// v16bf holds K = 16*(l>>4) + j.  Fragment index = ((t*4+kb)*32 + l)*16 + j.
__global__ __launch_bounds__(256) void k_split_w2(
    const float* __restrict__ W2, bf16_t* __restrict__ w2hi, bf16_t* __restrict__ w2lo)
{
    int idx = blockIdx.x * 256 + threadIdx.x;          // 0 .. 262143
    int j  = idx & 15;
    int l  = (idx >> 4) & 31;
    int kb = (idx >> 9) & 3;
    int t  = idx >> 11;                                // N-tile 0..127
    int Nc = l & 15;
    int K  = kb * 32 + (l >> 4) * 16 + j;              // 0..127
    float v = W2[(size_t)K * W2COLS + t * 16 + Nc];
    bf16_t hi = (bf16_t)v;
    w2hi[idx] = hi;
    w2lo[idx] = (bf16_t)(v - (float)hi);
}

// --- Zero/seed the global accumulators --------------------------------------
__global__ __launch_bounds__(256) void k_init(
    float* __restrict__ agg, float* __restrict__ s, unsigned* __restrict__ mkey)
{
    int i = blockIdx.x * 256 + threadIdx.x;
    if (i < N_NODES * V_OUT) agg[i] = 0.f;
    if (i < N_NODES * NHEAD) { s[i] = 0.f; mkey[i] = NEG_KEY; }
}

// --- Main fused edge kernel: one block = 16 edges, 8 waves ------------------
__global__ __launch_bounds__(256) void k_edge(
    const float* __restrict__ x, const int* __restrict__ eidx,
    const float* __restrict__ e, const float* __restrict__ W1,
    const float* __restrict__ b1, const float* __restrict__ b2,
    const float* __restrict__ att,
    const bf16_t* __restrict__ w2hi, const bf16_t* __restrict__ w2lo,
    float* __restrict__ xj_ws, float* __restrict__ alpha_ws)
{
    __shared__ float  eT[16][E_INF];
    __shared__ int    srcI[16], dstI[16];
    __shared__ bf16_t hHi[16][HID];
    __shared__ bf16_t hLo[16][HID];
    __shared__ float  xs[16][V_IN];
    __shared__ float  xd[16][V_IN];
    __shared__ float  b2s[W2COLS];
    __shared__ float  xjS[16][V_OUT];
    __shared__ float  xiS[16][V_OUT];

    const int tid = threadIdx.x;
    const int e0  = blockIdx.x * 16;

    if (tid < 16) {
        srcI[tid] = eidx[e0 + tid];
        dstI[tid] = eidx[N_EDGES + e0 + tid];
    }
    { int m = tid >> 4, c = tid & 15; eT[m][c] = e[(size_t)(e0 + m) * E_INF + c]; }
    for (int j = tid; j < W2COLS; j += 256) b2s[j] = b2[j];
    for (int j = tid; j < 16 * V_OUT; j += 256) { (&xjS[0][0])[j] = 0.f; (&xiS[0][0])[j] = 0.f; }
    __syncthreads();

    // gather endpoint features
    for (int idx = tid; idx < 16 * V_IN; idx += 256) {
        int m = idx >> 5, c = idx & 31;
        xs[m][c] = x[(size_t)srcI[m] * V_IN + c];
        xd[m][c] = x[(size_t)dstI[m] * V_IN + c];
    }
    // edge MLP layer 1 (tiny: fp32 VALU), split h into bf16 hi/lo
    for (int idx = tid; idx < 16 * HID; idx += 256) {
        int m = idx >> 7, k = idx & 127;
        float acc = b1[k];
#pragma unroll
        for (int c = 0; c < E_INF; ++c) acc = fmaf(eT[m][c], W1[c * HID + k], acc);
        acc = acc > 0.f ? acc : 0.f;
        bf16_t hi = (bf16_t)acc;
        hHi[m][k] = hi;
        hLo[m][k] = (bf16_t)(acc - (float)hi);
    }
    __syncthreads();

    const int lane = tid & 31;
    const int wid  = tid >> 5;

    // A fragments in contiguous k order, matching the B swizzle (not the ISA table order):
    // lane l, elem j -> M = l&15, K = j + 16*(l>>4)
    v16bf Ahi[4], Alo[4];
    {
        const int M = lane & 15;
        const int koff = (lane >> 4) * 16;
#pragma unroll
        for (int kb = 0; kb < 4; ++kb) {
#pragma unroll
            for (int j = 0; j < 16; ++j) {
                int kl = j + koff;
                Ahi[kb][j] = hHi[M][kb * 32 + kl];
                Alo[kb][j] = hLo[M][kb * 32 + kl];
            }
        }
    }

    v8f axj = {0, 0, 0, 0, 0, 0, 0, 0};
    v8f axi = {0, 0, 0, 0, 0, 0, 0, 0};
    const int nloc = lane & 15;
    const int mhi  = (lane >> 4) * 8;

    // 128 N-tiles over 8 waves; (t & 3) == (wid & 3) so each wave privately
    // accumulates one 16-wide o-group of xj/xi.
    for (int t = wid; t < 128; t += 8) {
        const int it = t >> 2;                    // input channel i = n/64
        v8f d = {0, 0, 0, 0, 0, 0, 0, 0};
        const bf16_t* bh = w2hi + ((size_t)t * 128 + lane) * 16;
        const bf16_t* bl = w2lo + ((size_t)t * 128 + lane) * 16;
#pragma unroll
        for (int kb = 0; kb < 4; ++kb) {
            v16bf Bhi = *(const v16bf*)(bh + (size_t)kb * 512);
            v16bf Blo = *(const v16bf*)(bl + (size_t)kb * 512);
            d = __builtin_amdgcn_wmma_f32_16x16x32_bf16(false, Ahi[kb], false, Bhi, (short)0, d, false, false);
            d = __builtin_amdgcn_wmma_f32_16x16x32_bf16(false, Ahi[kb], false, Blo, (short)0, d, false, false);
            d = __builtin_amdgcn_wmma_f32_16x16x32_bf16(false, Alo[kb], false, Bhi, (short)0, d, false, false);
        }
        // epilogue: w[e, it, o] = d + b2; fold into xj/xi immediately
        float b2v = b2s[t * 16 + nloc];
#pragma unroll
        for (int r = 0; r < 8; ++r) {
            int m = r + mhi;                      // edge row (C/D layout)
            float wv = d[r] + b2v;
            axj[r] = fmaf(xs[m][it], wv, axj[r]);
            axi[r] = fmaf(xd[m][it], wv, axi[r]);
        }
    }
    {
        const int og = wid & 3;                   // o-group (constant per wave)
#pragma unroll
        for (int r = 0; r < 8; ++r) {
            atomicAdd(&xjS[r + mhi][og * 16 + nloc], axj[r]);
            atomicAdd(&xiS[r + mhi][og * 16 + nloc], axi[r]);
        }
    }
    __syncthreads();

    // attention logits (pre-softmax) + leaky_relu(0.2)
    if (tid < 16 * NHEAD) {
        int m = tid >> 3, hh = tid & 7;
        float s1 = 0.f, s2 = 0.f;
#pragma unroll
        for (int c = 0; c < CH; ++c) {
            s1 = fmaf(xiS[m][hh * CH + c], att[hh * 2 * CH + c], s1);
            s2 = fmaf(xjS[m][hh * CH + c], att[hh * 2 * CH + CH + c], s2);
        }
        float a = s1 + s2;
        a = a > 0.f ? a : 0.2f * a;
        alpha_ws[(size_t)(e0 + m) * NHEAD + hh] = a;
    }
    // spill xj for the scatter pass
    for (int idx = tid; idx < 16 * V_OUT; idx += 256) {
        int m = idx >> 6, o = idx & 63;
        xj_ws[(size_t)(e0 + m) * V_OUT + o] = xjS[m][o];
    }
}

// --- segment max over incoming edges ----------------------------------------
__global__ __launch_bounds__(256) void k_max(
    const float* __restrict__ alpha, const int* __restrict__ eidx,
    unsigned* __restrict__ mkey)
{
    int idx = blockIdx.x * 256 + threadIdx.x;
    if (idx >= N_EDGES * NHEAD) return;
    int eN = idx >> 3, h = idx & 7;
    int dst = eidx[N_EDGES + eN];
    atomicMax(&mkey[dst * NHEAD + h], fkey(alpha[idx]));
}

// --- exp(alpha - max) and segment sum (alpha overwritten in place) ----------
__global__ __launch_bounds__(256) void k_expsum(
    float* __restrict__ alpha, const int* __restrict__ eidx,
    const unsigned* __restrict__ mkey, float* __restrict__ s)
{
    int idx = blockIdx.x * 256 + threadIdx.x;
    if (idx >= N_EDGES * NHEAD) return;
    int eN = idx >> 3, h = idx & 7;
    int dst = eidx[N_EDGES + eN];
    float m = kinv(mkey[dst * NHEAD + h]);
    float ea = expf(alpha[idx] - m);
    alpha[idx] = ea;
    atomicAdd(&s[dst * NHEAD + h], ea);
}

// --- attended scatter-add (includes per-edge bias, as in reference) ---------
__global__ __launch_bounds__(256) void k_scatter(
    const float* __restrict__ xj_ws, const float* __restrict__ ea,
    const float* __restrict__ s, const int* __restrict__ eidx,
    const float* __restrict__ bias, float* __restrict__ agg)
{
    int idx = blockIdx.x * 256 + threadIdx.x;
    if (idx >= N_EDGES * V_OUT) return;
    int eN = idx >> 6, o = idx & 63, h = o >> 3;
    int dst = eidx[N_EDGES + eN];
    float a = ea[eN * NHEAD + h] / (s[dst * NHEAD + h] + 1e-16f);
    atomicAdd(&agg[dst * V_OUT + o], fmaf(xj_ws[idx], a, bias[o]));
}

// --- root transform + bias + ELU, single write of d_out ---------------------
__global__ __launch_bounds__(256) void k_final(
    const float* __restrict__ agg, const float* __restrict__ x,
    const float* __restrict__ root, const float* __restrict__ bias,
    float* __restrict__ out)
{
    int idx = blockIdx.x * 256 + threadIdx.x;
    if (idx >= N_NODES * V_OUT) return;
    int n = idx >> 6, o = idx & 63;
    float acc = agg[idx] + bias[o];
#pragma unroll
    for (int i = 0; i < V_IN; ++i) acc = fmaf(x[n * V_IN + i], root[i * V_OUT + o], acc);
    out[idx] = acc > 0.f ? acc : expm1f(acc);
}

extern "C" void kernel_launch(void* const* d_in, const int* in_sizes, int n_in,
                              void* d_out, int out_size, void* d_ws, size_t ws_size,
                              hipStream_t stream)
{
    const float* x    = (const float*)d_in[0];
    const int*   eidx = (const int*)  d_in[1];
    const float* e    = (const float*)d_in[2];
    const float* W1   = (const float*)d_in[3];
    const float* b1   = (const float*)d_in[4];
    const float* W2   = (const float*)d_in[5];
    const float* b2   = (const float*)d_in[6];
    const float* att  = (const float*)d_in[7];
    const float* bias = (const float*)d_in[8];
    const float* root = (const float*)d_in[9];
    float* out = (float*)d_out;

    char* ws = (char*)d_ws;
    bf16_t*   w2hi  = (bf16_t*)(ws + OFF_W2HI);
    bf16_t*   w2lo  = (bf16_t*)(ws + OFF_W2LO);
    float*    xj_ws = (float*)(ws + OFF_XJ);
    float*    alpha = (float*)(ws + OFF_ALPHA);
    unsigned* mkey  = (unsigned*)(ws + OFF_MKEY);
    float*    ssum  = (float*)(ws + OFF_S);
    float*    agg   = (float*)(ws + OFF_AGG);

    // 1) W2 -> bf16 hi/lo, pre-swizzled to WMMA B-fragment order (1 MB total)
    k_split_w2<<<262144 / 256, 256, 0, stream>>>(W2, w2hi, w2lo);
    // 2) zero/seed accumulators
    k_init<<<(N_NODES * V_OUT + 255) / 256, 256, 0, stream>>>(agg, ssum, mkey);
    // 3) fused edge pipeline (edge MLP + WMMA GEMM + xj/xi + attention logits)
    k_edge<<<N_EDGES / 16, 256, 0, stream>>>(x, eidx, e, W1, b1, b2, att,
                                             w2hi, w2lo, xj_ws, alpha);
    // 4) segment softmax
    k_max   <<<(N_EDGES * NHEAD + 255) / 256, 256, 0, stream>>>(alpha, eidx, mkey);
    k_expsum<<<(N_EDGES * NHEAD + 255) / 256, 256, 0, stream>>>(alpha, eidx, mkey, ssum);
    // 5) attended scatter-add
    k_scatter<<<(N_EDGES * V_OUT + 255) / 256, 256, 0, stream>>>(xj_ws, alpha, ssum,
                                                                 eidx, bias, agg);
    // 6) root weight + bias + ELU
    k_final<<<(N_NODES * V_OUT + 255) / 256, 256, 0, stream>>>(agg, x, root, bias, out);
}
